// Merge_75376676045416
// MI455X (gfx1250) — compile-verified, matrix operand order checked by reading, not run on hardware
//
#include <hip/hip_runtime.h>
#include <hip/hip_bf16.h>

// MI455X (gfx1250).
// Hot path: fused (zn @ zn^T) + per-row top-6 with v_wmma_f32_16x16x32_bf16.
// zn (16 MB bf16) is L2-resident (192 MB L2) -> N^2*512 GEMM is compute bound.
// A fragments live in VGPRs (loop-invariant per wave); B panels double-buffer
// through LDS via global_load_async_to_lds_b128 + s_wait_asynccnt 0x4; the
// K loop runs two independent WMMA accumulator chains to avoid serializing on
// one D-register chain and one B-register set.

typedef __attribute__((ext_vector_type(16))) __bf16 bf16x16;
typedef __attribute__((ext_vector_type(4)))  __bf16 bf16x4;
typedef __attribute__((ext_vector_type(8)))  float  f32x8;

#define HID 256
#define ZDIM 512   // 2*HIDDEN
#define SEQ 200
#define XW  400    // 2*SEQ
#define KNN 5

// 16-element bf16 WMMA fragment (CDNA5 16-bit A/B layout, lane: m=lane&15, h=lane>>4):
//   elements 0..7  <-> k = base + 8h .. +7        (one b128)
//   elements 8..15 <-> k = base + 16 + 8h .. +7   (one b128)
__device__ __forceinline__ void load_frag(const __bf16* p, bf16x16& f) {
    uint4 lo = *reinterpret_cast<const uint4*>(p);
    uint4 hi = *reinterpret_cast<const uint4*>(p + 16);
    uint4* d = reinterpret_cast<uint4*>(&f);
    d[0] = lo; d[1] = hi;
}

// float4 -> 4 bf16 (v_cvt_pk_bf16_f32 x2) -> one b64 LDS store
__device__ __forceinline__ void cvt_store4(__bf16* dst, float4 f) {
    bf16x4 h;
    h[0] = (__bf16)f.x; h[1] = (__bf16)f.y; h[2] = (__bf16)f.z; h[3] = (__bf16)f.w;
    *reinterpret_cast<uint2*>(dst) = *reinterpret_cast<uint2*>(&h);
}

// Issue the 16 KB B panel as 4 x b128-per-lane async LDS copies.
__device__ __forceinline__ void async_panel(const __bf16* gsrc, unsigned ldsbase, int tid) {
    unsigned long long g = (unsigned long long)gsrc + (unsigned long long)tid * 16ull;
    unsigned l = ldsbase + (unsigned)tid * 16u;
#pragma unroll
    for (int q = 0; q < 4; ++q)
        asm volatile("global_load_async_to_lds_b128 %0, %1, off"
                     :: "v"(l + q * 4096u), "v"(g + (unsigned long long)q * 4096ull)
                     : "memory");
}

// out[m][n] = act( sum_k A[m][k]*W[n][k] + bias[n] )  i.e. A @ W^T.
// WG = 8 waves = 2x4 tile grid = 32x64 output tile. K compile-time; each thread
// has a fixed staging slot with hoisted pointers.
template<int K, int RELU>
__global__ void __launch_bounds__(256, 2)
gemm_bias_act(const float* __restrict__ A, const float* __restrict__ W,
              const float* __restrict__ bias, float* __restrict__ out, int Nc) {
    __shared__ __align__(16) __bf16 As[32 * 32];
    __shared__ __align__(16) __bf16 Bs[64 * 32];
    const int tid  = threadIdx.x;
    const int lane = tid & 31, w = tid >> 5;
    const int wr = w >> 2, wc = w & 3;
    const int m15 = lane & 15, hh = lane >> 4;
    const int rowbase = blockIdx.x * 32;
    const int colbase = blockIdx.y * 64;

    // fixed staging slots: 4 consecutive k per thread
    const int rs = tid >> 3, ks = (tid & 7) * 4;
    const float* aptr  = A + (size_t)(rowbase + rs) * K + ks;
    const float* bptr0 = W + (size_t)(colbase + rs) * K + ks;
    const float* bptr1 = W + (size_t)(colbase + rs + 32) * K + ks;
    __bf16* asd  = &As[rs * 32 + ks];
    __bf16* bsd0 = &Bs[rs * 32 + ks];
    __bf16* bsd1 = &Bs[(rs + 32) * 32 + ks];

    f32x8 c = {};
    for (int kc = 0; kc < K; kc += 32) {
        float4 fa = *reinterpret_cast<const float4*>(aptr);
        float4 f0 = *reinterpret_cast<const float4*>(bptr0);
        float4 f1 = *reinterpret_cast<const float4*>(bptr1);
        aptr += 32; bptr0 += 32; bptr1 += 32;
        __syncthreads();                     // previous tile consumed
        cvt_store4(asd, fa); cvt_store4(bsd0, f0); cvt_store4(bsd1, f1);
        __syncthreads();
        bf16x16 a, b;
        load_frag(&As[(wr * 16 + m15) * 32 + 8 * hh], a);
        load_frag(&Bs[(wc * 16 + m15) * 32 + 8 * hh], b);
        c = __builtin_amdgcn_wmma_f32_16x16x32_bf16(false, a, false, b, (short)0, c, false, false);
    }
    const int ng = colbase + wc * 16 + m15;
    const float bv = bias[ng];
#pragma unroll
    for (int r = 0; r < 8; ++r) {
        int mg = rowbase + wr * 16 + r + 8 * hh;
        float v = c[r] + bv;
        if (RELU) v = fmaxf(v, 0.0f);
        out[(size_t)mg * Nc + ng] = v;
    }
}

// z_sim[i] = dot(zp[i], zp[i+1 mod N]); z_sim[N-1] = -1e8. One wave per row,
// lane owns 8 consecutive dims (two b128 loads per operand row).
__global__ void zsim_kernel(const float* __restrict__ zp, float* __restrict__ zs, int N) {
    int w = threadIdx.x >> 5, lane = threadIdx.x & 31;
    int row = blockIdx.x * 8 + w;
    int nrow = (row + 1 == N) ? 0 : row + 1;
    const float4* pa = reinterpret_cast<const float4*>(zp + (size_t)row * HID + lane * 8);
    const float4* pb = reinterpret_cast<const float4*>(zp + (size_t)nrow * HID + lane * 8);
    float4 a0 = pa[0], a1 = pa[1], b0 = pb[0], b1 = pb[1];
    float s = a0.x * b0.x + a0.y * b0.y + a0.z * b0.z + a0.w * b0.w
            + a1.x * b1.x + a1.y * b1.y + a1.z * b1.z + a1.w * b1.w;
#pragma unroll
    for (int d = 16; d > 0; d >>= 1) s += __shfl_xor(s, d, 32);
    if (lane == 0) zs[row] = (row == N - 1) ? -1.0e8f : s;
}

// indices_comp membership: i in comp iff ascending-rank(z_sim[i], ties by index) < N/2.
__global__ void comp_kernel(const float* __restrict__ zs, int* __restrict__ comp, int N) {
    __shared__ __align__(16) float s[2048];
    int i = blockIdx.x * blockDim.x + threadIdx.x;
    float zi = zs[i];
    int rank = 0;
    for (int cb = 0; cb < N; cb += 2048) {
        __syncthreads();
        for (int t = threadIdx.x; t < 512; t += 256)
            reinterpret_cast<float4*>(s)[t] = reinterpret_cast<const float4*>(zs + cb)[t];
        __syncthreads();
        const float4* s4 = reinterpret_cast<const float4*>(s);
        for (int t = 0; t < 512; ++t) {
            float4 v = s4[t];
            int j = cb + t * 4;
            rank += (v.x < zi || (v.x == zi && j + 0 < i)) ? 1 : 0;
            rank += (v.y < zi || (v.y == zi && j + 1 < i)) ? 1 : 0;
            rank += (v.z < zi || (v.z == zi && j + 2 < i)) ? 1 : 0;
            rank += (v.w < zi || (v.w == zi && j + 3 < i)) ? 1 : 0;
        }
    }
    comp[i] = (rank < (N >> 1)) ? 1 : 0;
}

// z_merge -> normalized bf16 zn, plus y_merge. One wave per row; lane owns 16
// consecutive dims (4x b128 loads per operand, 4x b64 packed bf16 stores).
__global__ void merge_zn_y(const float* __restrict__ z, const float* __restrict__ y,
                           const int* __restrict__ comp, __bf16* __restrict__ znb,
                           float* __restrict__ ymerge, int N) {
    int w = threadIdx.x >> 5, lane = threadIdx.x & 31;
    int row = blockIdx.x * 8 + w;
    int nrow = (row + 1 == N) ? 0 : row + 1;
    int c = comp[row];
    const int k0 = lane * 16;
    const float4* pa = reinterpret_cast<const float4*>(z + (size_t)row * ZDIM + k0);
    const float4* pb = reinterpret_cast<const float4*>(z + (size_t)nrow * ZDIM + k0);
    float m[16];
    float ss = 0.f;
#pragma unroll
    for (int q = 0; q < 4; ++q) {
        float4 a = pa[q], b = pb[q];
        float4 v;
        v.x = c ? a.x : 0.5f * (a.x + b.x);
        v.y = c ? a.y : 0.5f * (a.y + b.y);
        v.z = c ? a.z : 0.5f * (a.z + b.z);
        v.w = c ? a.w : 0.5f * (a.w + b.w);
        m[q * 4 + 0] = v.x; m[q * 4 + 1] = v.y; m[q * 4 + 2] = v.z; m[q * 4 + 3] = v.w;
        ss += v.x * v.x + v.y * v.y + v.z * v.z + v.w * v.w;
    }
#pragma unroll
    for (int d = 16; d > 0; d >>= 1) ss += __shfl_xor(ss, d, 32);
    float inv = 1.0f / fmaxf(sqrtf(ss), 1e-12f);
#pragma unroll
    for (int q = 0; q < 4; ++q) {
        bf16x4 h;
        h[0] = (__bf16)(m[q * 4 + 0] * inv); h[1] = (__bf16)(m[q * 4 + 1] * inv);
        h[2] = (__bf16)(m[q * 4 + 2] * inv); h[3] = (__bf16)(m[q * 4 + 3] * inv);
        *reinterpret_cast<uint2*>(znb + (size_t)row * ZDIM + k0 + q * 4) =
            *reinterpret_cast<uint2*>(&h);
    }
    if (lane == 0) ymerge[row] = c ? y[row] : fminf(y[row], y[nrow]);
}

// Stable partition of the 400-wide merged row by mask (ones first, order kept),
// replacing the reference's stable argsort. One wave per row, ballot prefix sums.
__global__ void xmerge_kernel(const float* __restrict__ x, const float* __restrict__ xm,
                              const int* __restrict__ comp,
                              float* __restrict__ ox1, float* __restrict__ ox2,
                              float* __restrict__ om, int N) {
    int w = threadIdx.x >> 5, lane = threadIdx.x & 31;
    int row = blockIdx.x * 8 + w;
    int nrow = (row + 1 == N) ? 0 : row + 1;
    int c = comp[row];
    float xv[13];
    unsigned bal[13];
    int total_ones = 0;
#pragma unroll
    for (int cc = 0; cc < 13; ++cc) {
        int j = cc * 32 + lane;
        bool valid = j < XW;
        float xval = 0.f, mval = 0.f;
        if (valid) {
            if (j < SEQ) { xval = x[(size_t)row * SEQ + j];  mval = xm[(size_t)row * SEQ + j]; }
            else {
                int j2 = j - SEQ;
                xval = x[(size_t)nrow * SEQ + j2];
                mval = c ? 0.f : xm[(size_t)nrow * SEQ + j2];   // comp rows: zero 2nd-half mask
            }
        }
        unsigned b = __builtin_amdgcn_ballot_w32(valid && (mval != 0.0f));
        xv[cc] = xval; bal[cc] = b;
        total_ones += __popc((int)b);
    }
    int ones_pre = 0, zeros_pre = 0;
#pragma unroll
    for (int cc = 0; cc < 13; ++cc) {
        int j = cc * 32 + lane;
        unsigned vmask = (cc < 12) ? 0xffffffffu : 0x0000ffffu;  // last chunk: 16 valid
        unsigned lm = (1u << lane) - 1u;
        unsigned b = bal[cc];
        int mself = (b >> lane) & 1;
        int pos = mself ? (ones_pre + __popc((int)(b & lm)))
                        : (total_ones + zeros_pre + __popc((int)(~b & vmask & lm)));
        if (j < XW) {
            size_t o = (size_t)row * XW + pos;
            ox1[o] = xv[cc]; ox2[o] = xv[cc];
            om[o] = mself ? 1.0f : 0.0f;
        }
        ones_pre  += __popc((int)b);
        zeros_pre += __popc((int)(~b & vmask));
    }
}

// Fused adjacency GEMM + running top-6 per row.
// WG = 8 waves x 16 rows = 128 rows. A fragments preloaded to VGPRs once
// (launch_bounds(256,1) -> no spills). B panels double-buffer through LDS via
// async copies; two accumulator chains (even/odd K-step) keep the XDL pipe fed.
__global__ void __launch_bounds__(256, 1)
adj_topk(const __bf16* __restrict__ zn, int N,
         float* __restrict__ ei0, float* __restrict__ ei1, float* __restrict__ ea) {
    __shared__ __align__(16) __bf16 Bs[2][16 * ZDIM];   // 2 x 16 KB
    __shared__ float Cs[8 * 256];                       // 8 KB
    const int tid  = threadIdx.x;
    const int lane = tid & 31, w = tid >> 5;
    const int m15 = lane & 15, hh = lane >> 4;
    const int rowbase = blockIdx.x * 128;

    // Preload this wave's 16 A fragments (rows rowbase + w*16 .. +15, K=512).
    bf16x16 afrag[16];
    const __bf16* arow = zn + (size_t)(rowbase + w * 16 + m15) * ZDIM;
#pragma unroll
    for (int ksi = 0; ksi < 16; ++ksi)
        load_frag(arow + ksi * 32 + 8 * hh, afrag[ksi]);

    float tval[KNN + 1];
    int   tidx[KNN + 1];
#pragma unroll
    for (int t = 0; t < KNN + 1; ++t) { tval[t] = -3.4e38f; tidx[t] = 0x7fffffff; }

    // LDS byte offset of Bs (generic LDS address low 32 bits == LDS offset).
    const unsigned bs_base = (unsigned)(uintptr_t)(&Bs[0][0]);
    const int nct = N >> 4;

    async_panel(zn, bs_base, tid);   // panel 0 -> buffer 0
    for (int ct = 0; ct < nct; ++ct) {
        if (ct + 1 < nct) {
            // start next panel into the other buffer, then wait for *current*
            // panel only (async ops complete in order: <=4 left => ct's 4 done)
            async_panel(zn + (size_t)(ct + 1) * 16 * ZDIM,
                        bs_base + ((unsigned)(ct + 1) & 1u) * 16384u, tid);
            if (ct + 2 < nct)
                __builtin_prefetch(zn + (size_t)(ct + 2) * 16 * ZDIM + tid * 32, 0, 1);
            asm volatile("s_wait_asynccnt 0x4" ::: "memory");
        } else {
            asm volatile("s_wait_asynccnt 0x0" ::: "memory");
        }
        __syncthreads();   // every wave's current panel resident

        const __bf16* bb = &Bs[ct & 1][0];
        // Two independent accumulator chains (even/odd K-step) with separate
        // B fragment registers -> B loads overlap the other chain's WMMA and
        // dependent WMMAs don't serialize on one D chain.
        f32x8 c0 = {}, c1 = {};
#pragma unroll
        for (int kp = 0; kp < 8; ++kp) {
            bf16x16 b0, b1;
            load_frag(&bb[m15 * ZDIM + (2 * kp + 0) * 32 + 8 * hh], b0);
            load_frag(&bb[m15 * ZDIM + (2 * kp + 1) * 32 + 8 * hh], b1);
            c0 = __builtin_amdgcn_wmma_f32_16x16x32_bf16(false, afrag[2 * kp + 0], false, b0,
                                                         (short)0, c0, false, false);
            c1 = __builtin_amdgcn_wmma_f32_16x16x32_bf16(false, afrag[2 * kp + 1], false, b1,
                                                         (short)0, c1, false, false);
        }
        // spill 16x16 tile through per-wave LDS scratch so one lane owns one row
#pragma unroll
        for (int r = 0; r < 8; ++r)
            Cs[w * 256 + (r + 8 * hh) * 16 + m15] = c0[r] + c1[r];
        __syncthreads();

        if (lane < 16) {
#pragma unroll
            for (int t = 0; t < 16; ++t) {
                float v = Cs[w * 256 + lane * 16 + t];
                int id = ct * 16 + t;
                // lax.top_k order: descending value, ties -> lower index first
                bool beat = (v > tval[KNN]) || ((v == tval[KNN]) && (id < tidx[KNN]));
                if (beat) {
                    tval[KNN] = v; tidx[KNN] = id;
#pragma unroll
                    for (int s = KNN; s > 0; --s) {
                        bool up = (tval[s] > tval[s - 1]) ||
                                  ((tval[s] == tval[s - 1]) && (tidx[s] < tidx[s - 1]));
                        if (up) {
                            float fv = tval[s]; int fi = tidx[s];
                            tval[s] = tval[s - 1]; tidx[s] = tidx[s - 1];
                            tval[s - 1] = fv;      tidx[s - 1] = fi;
                        }
                    }
                }
            }
        }
    }
    if (lane < 16) {
        int row = rowbase + w * 16 + lane;
#pragma unroll
        for (int j = 1; j <= KNN; ++j) {   // drop slot 0 (self / max)
            size_t o = (size_t)row * KNN + (j - 1);
            ei0[o] = (float)row;
            ei1[o] = (float)tidx[j];
            ea[o]  = tval[j];
        }
    }
}

__global__ void copy_temporal(const int* __restrict__ tei, const float* __restrict__ tea,
                              float* __restrict__ otei, float* __restrict__ otea,
                              int nei, int nea) {
    int i = blockIdx.x * blockDim.x + threadIdx.x;
    if (i < nei) otei[i] = (float)tei[i];
    if (i < nea) otea[i] = tea[i];
}

extern "C" void kernel_launch(void* const* d_in, const int* in_sizes, int n_in,
                              void* d_out, int out_size, void* d_ws, size_t ws_size,
                              hipStream_t stream) {
    (void)n_in; (void)out_size; (void)ws_size;
    const float* z   = (const float*)d_in[0];
    const float* x   = (const float*)d_in[1];
    const float* y   = (const float*)d_in[2];
    const float* xmk = (const float*)d_in[3];
    const int*   tei = (const int*)d_in[4];
    const float* tea = (const float*)d_in[5];
    const float* W1  = (const float*)d_in[6];
    const float* b1  = (const float*)d_in[7];
    const float* W2  = (const float*)d_in[8];
    const float* b2  = (const float*)d_in[9];
    const int N = in_sizes[2];      // 16384
    const int E = in_sizes[5];      // 32768

    // workspace layout
    char* ws = (char*)d_ws;
    float* h    = (float*)ws;                          // N*256 f32
    float* zp   = h + (size_t)N * HID;                 // N*256 f32
    float* zs   = zp + (size_t)N * HID;                // N f32
    int*   comp = (int*)(zs + N);                      // N i32
    __bf16* znb = (__bf16*)(comp + N);                 // N*512 bf16

    // output layout (flattened tuple, return order)
    float* out    = (float*)d_out;
    const size_t LEN_X = (size_t)N * XW;
    float* o_semx = out;
    float* o_ei0  = o_semx + LEN_X;
    float* o_ei1  = o_ei0 + (size_t)N * KNN;
    float* o_ea   = o_ei1 + (size_t)N * KNN;
    float* o_y    = o_ea  + (size_t)N * KNN;
    float* o_tmpx = o_y + N;
    float* o_tei  = o_tmpx + LEN_X;
    float* o_tea  = o_tei + (size_t)2 * E;
    float* o_mask = o_tea + E;

    // 1) h = relu(z @ W1^T + b1)      (K=512)
    gemm_bias_act<2 * HID, 1><<<dim3(N / 32, HID / 64), 256, 0, stream>>>(z, W1, b1, h, HID);
    // 2) zp = h @ W2^T + b2           (K=256)
    gemm_bias_act<HID, 0><<<dim3(N / 32, HID / 64), 256, 0, stream>>>(h, W2, b2, zp, HID);
    // 3) neighbor-pair similarity
    zsim_kernel<<<N / 8, 256, 0, stream>>>(zp, zs, N);
    // 4) indices_comp membership by rank
    comp_kernel<<<N / 256, 256, 0, stream>>>(zs, comp, N);
    // 5) z_merge -> normalized bf16 zn, and y_merge
    merge_zn_y<<<N / 8, 256, 0, stream>>>(z, y, comp, znb, o_y, N);
    // 6) x_merge / mask stable partition (writes semantic x, temporal x, mask)
    xmerge_kernel<<<N / 8, 256, 0, stream>>>(x, xmk, comp, o_semx, o_tmpx, o_mask, N);
    // 7) fused zn@zn^T + top-6 -> edges
    adj_topk<<<N / 128, 256, 0, stream>>>(znb, N, o_ei0, o_ei1, o_ea);
    // 8) temporal passthrough
    copy_temporal<<<(2 * E + 255) / 256, 256, 0, stream>>>(tei, tea, o_tei, o_tea, 2 * E, E);
}
